// Encoder_79061757985215
// MI455X (gfx1250) — compile-verified, matrix operand order checked by reading, not run on hardware
//
#include <hip/hip_runtime.h>
#include <hip/hip_bf16.h>
#include <stddef.h>

// ---------------------------------------------------------------------------
// Types for CDNA5 WMMA (wave32, 16x16x32 bf16 -> f32)
// ---------------------------------------------------------------------------
typedef __attribute__((ext_vector_type(16))) __bf16    v16bf;
typedef __attribute__((ext_vector_type(8)))  float     v8f;
typedef __attribute__((ext_vector_type(4)))  unsigned  u32x4;

union Frag16 {
    v16bf bf;
    u32x4 q[2];
    unsigned short us[16];
};

__device__ __forceinline__ v8f vzero8() {
    v8f z = {0.f, 0.f, 0.f, 0.f, 0.f, 0.f, 0.f, 0.f};
    return z;
}

__device__ __forceinline__ v8f wmma_bf16(v16bf a, v16bf b, v8f c) {
    // (neg_a, A, neg_b, B, c_mod, C, reuse_a, reuse_b)
    return __builtin_amdgcn_wmma_f32_16x16x32_bf16(false, a, false, b,
                                                   (short)0, c, false, false);
}

__device__ __forceinline__ unsigned short f2bf(float f) {
    unsigned u = __float_as_uint(f);
    unsigned r = 0x7fffu + ((u >> 16) & 1u);   // round-to-nearest-even
    return (unsigned short)((u + r) >> 16);
}
__device__ __forceinline__ float bf2f(unsigned short s) {
    return __uint_as_float(((unsigned)s) << 16);
}

// Fast gate functions on the serial chain: v_exp_f32 / v_rcp_f32 are TRANS32
// ops that co-execute with WMMA on CDNA5.
__device__ __forceinline__ float fast_sigmoid(float x) {
    return __builtin_amdgcn_rcpf(1.f + __expf(-x));
}
__device__ __forceinline__ float fast_tanh(float x) {
    const float t = __expf(-2.f * __builtin_fabsf(x));   // in (0, 1]
    const float r = (1.f - t) * __builtin_amdgcn_rcpf(1.f + t);
    return __builtin_copysignf(r, x);
}

// 16-bit A/B fragment (16x32 MxK or 32x16 KxN from transposed storage).
// src is row-major with leading dim ldk (elements); lane L%16 selects row,
// lane bit4 selects K-chunk base {0,8}; halves 8..15 are K+16.
__device__ __forceinline__ v16bf ldfrag(const unsigned short* base, int row0,
                                        int ldk, int k0, int lane) {
    const int r  = row0 + (lane & 15);
    const int kb = k0 + ((lane & 16) ? 8 : 0);
    const unsigned short* p = base + (size_t)r * (size_t)ldk + kb;
    Frag16 f;
    f.q[0] = *(const u32x4*)(p);
    f.q[1] = *(const u32x4*)(p + 16);
    return f.bf;
}

// ---------------------------------------------------------------------------
// Problem constants
// ---------------------------------------------------------------------------
#define BATCH 32
#define TLEN  2048
#define UNITS 256
#define N3U   768          // 3*U
#define MROWS (BATCH * TLEN)   // 65536
#define OUT_MAIN (BATCH * TLEN * 2 * UNITS)   // 33554432 floats
#define LDH 264            // padded LDS row stride (halves) to avoid bank conflicts

// ---------------------------------------------------------------------------
// zero fp32 buffer
// ---------------------------------------------------------------------------
__global__ void zero_f32_kernel(float* __restrict__ p, int n) {
    int i = blockIdx.x * blockDim.x + threadIdx.x;
    if (i < n) p[i] = 0.f;
}

// ---------------------------------------------------------------------------
// transpose + convert weights: src fp32 [K, 768] -> dst bf16 [768, K]
// ---------------------------------------------------------------------------
__global__ void transpose_bf16_kernel(const float* __restrict__ src,
                                      unsigned short* __restrict__ dst, int K) {
    int idx = blockIdx.x * blockDim.x + threadIdx.x;
    if (idx >= N3U * K) return;
    int n = idx / K;
    int k = idx - n * K;
    dst[idx] = f2bf(src[k * N3U + n]);
}

// ---------------------------------------------------------------------------
// Layer-0 input projection (in_dim == 1): xp[bt,c] = x[bt]*Wk[c] + b0[c]
// Produces both directions' projections (bf16).
// ---------------------------------------------------------------------------
__global__ void proj0_kernel(const float* __restrict__ x,
                             const float* __restrict__ wkf,
                             const float* __restrict__ bf0,
                             const float* __restrict__ wkb,
                             const float* __restrict__ bb0,
                             unsigned short* __restrict__ xpf,
                             unsigned short* __restrict__ xpb) {
    int idx = blockIdx.x * blockDim.x + threadIdx.x;
    if (idx >= MROWS * N3U) return;
    int bt = idx / N3U;
    int c  = idx - bt * N3U;
    float xv = x[bt];
    xpf[idx] = f2bf(xv * wkf[c] + bf0[c]);
    xpb[idx] = f2bf(xv * wkb[c] + bb0[c]);
}

// ---------------------------------------------------------------------------
// Input projection GEMM for layers 1,2:
//   X  bf16 [65536, 512]   (previous layer's bidirectional output)
//   WT bf16 [768, 512]     (Wk transposed)
//   out bf16 [65536, 768] = X @ Wk + bias(row0)
// One wave computes a 32x32 tile (2x2 WMMA), 8 waves per block.
// ---------------------------------------------------------------------------
__global__ __launch_bounds__(256) void proj_gemm_kernel(
        const unsigned short* __restrict__ X,
        const unsigned short* __restrict__ WT,
        const float* __restrict__ bias,
        unsigned short* __restrict__ out) {
    const int K = 2 * UNITS;       // 512
    const int lane = threadIdx.x & 31;
    const int gw = blockIdx.x * 8 + (threadIdx.x >> 5);
    const int tn = gw % (N3U / 32);
    const int tm = gw / (N3U / 32);
    const int m0 = tm * 32, n0 = tn * 32;

    v8f C00 = vzero8(), C01 = vzero8(), C10 = vzero8(), C11 = vzero8();

#pragma unroll 4
    for (int k0 = 0; k0 < K; k0 += 32) {
        v16bf a0 = ldfrag(X, m0,      K, k0, lane);
        v16bf a1 = ldfrag(X, m0 + 16, K, k0, lane);
        v16bf b0 = ldfrag(WT, n0,      K, k0, lane);
        v16bf b1 = ldfrag(WT, n0 + 16, K, k0, lane);
        C00 = wmma_bf16(a0, b0, C00);
        C01 = wmma_bf16(a0, b1, C01);
        C10 = wmma_bf16(a1, b0, C10);
        C11 = wmma_bf16(a1, b1, C11);
    }

    const int hi8 = (lane & 16) ? 8 : 0;
    const int ln  = lane & 15;
    v8f Ct[2][2] = {{C00, C01}, {C10, C11}};
#pragma unroll
    for (int ct = 0; ct < 2; ++ct) {
#pragma unroll
        for (int cn = 0; cn < 2; ++cn) {
            const int n = n0 + cn * 16 + ln;
            const float bs = bias[n];
#pragma unroll
            for (int v = 0; v < 8; ++v) {
                const int m = m0 + ct * 16 + v + hi8;
                out[(size_t)m * N3U + n] = f2bf(Ct[ct][cn][v] + bs);
            }
        }
    }
}

// ---------------------------------------------------------------------------
// Persistent recurrent scan. grid = 2 blocks (dir 0 = fw, 1 = bw),
// 512 threads = 16 waves. Wave w owns hidden units [16w, 16w+16) and keeps
// its slice of WrT (3 gate column groups x 8 K-steps) resident in VGPRs.
// h is double-buffered in LDS as bf16 for the WMMA A-operand; the fp32 state
// lives in the owning wave's registers (exactly the C-tile layout).
// ---------------------------------------------------------------------------
__global__ __launch_bounds__(512) void gru_scan_kernel(
        const unsigned short* __restrict__ xp_fw,
        const unsigned short* __restrict__ xp_bw,
        const unsigned short* __restrict__ wrt_fw,
        const unsigned short* __restrict__ wrt_bw,
        const float* __restrict__ b1_fw,   // recurrent bias row [768]
        const float* __restrict__ b1_bw,
        const float* __restrict__ h0,      // [2][32][256]
        float* __restrict__ hfin,          // [2][32][256]
        void* __restrict__ outp,           // [32][2048][512] bf16 or f32
        int out_f32,
        float* __restrict__ out_states) {  // nullable, [32][512]
    __shared__ unsigned short hbf[2][32 * LDH];

    const int dir = blockIdx.x;
    const unsigned short* xp  = dir ? xp_bw  : xp_fw;
    const unsigned short* wrt = dir ? wrt_bw : wrt_fw;
    const float* b1  = dir ? b1_bw : b1_fw;
    const float* h0d = h0 + dir * (BATCH * UNITS);
    float* hfd       = hfin + dir * (BATCH * UNITS);

    const int tid  = threadIdx.x;
    const int lane = tid & 31;
    const int w    = tid >> 5;
    const int jb   = w * 16;                 // hidden-unit base for this wave
    const int ln   = lane & 15;
    const int hi8  = (lane & 16) ? 8 : 0;

    // ---- Persistent Wr fragments: Bfr[k-step][gate] ----
    v16bf Bfr[8][3];
#pragma unroll
    for (int k = 0; k < 8; ++k)
#pragma unroll
        for (int g = 0; g < 3; ++g)
            Bfr[k][g] = ldfrag(wrt, g * UNITS + jb, UNITS, k * 32, lane);

    const float bz = b1[jb + ln];
    const float br = b1[UNITS + jb + ln];
    const float bh = b1[2 * UNITS + jb + ln];

    // fp32 state held in registers: hprev[mt][v] = h(b = 16*mt + v + hi8, jb+ln)
    float hprev[2][8];
#pragma unroll
    for (int mt = 0; mt < 2; ++mt)
#pragma unroll
        for (int v = 0; v < 8; ++v)
            hprev[mt][v] = h0d[(mt * 16 + v + hi8) * UNITS + jb + ln];

    // cooperative fill of hbf[0] with bf16(h0)
    for (int e = tid; e < BATCH * UNITS; e += 512)
        hbf[0][(e >> 8) * LDH + (e & 255)] = f2bf(h0d[e]);
    __syncthreads();

    for (int t = 0; t < TLEN; ++t) {
        const int tt  = dir ? (TLEN - 1 - t) : t;
        const int cur = t & 1;
        const int nxt = cur ^ 1;

        v8f C[2][3];
#pragma unroll
        for (int mt = 0; mt < 2; ++mt)
#pragma unroll
            for (int g = 0; g < 3; ++g)
                C[mt][g] = vzero8();

        // hp = h @ Wr  (K = 256, 8 k-steps; A from LDS, B from registers)
#pragma unroll
        for (int k = 0; k < 8; ++k) {
#pragma unroll
            for (int mt = 0; mt < 2; ++mt) {
                const unsigned short* ap =
                    &hbf[cur][(mt * 16 + ln) * LDH + k * 32 + hi8];
                Frag16 a;
                a.q[0] = *(const u32x4*)(ap);
                a.q[1] = *(const u32x4*)(ap + 16);
#pragma unroll
                for (int g = 0; g < 3; ++g)
                    C[mt][g] = wmma_bf16(a.bf, Bfr[k][g], C[mt][g]);
            }
        }

        // gates + state update (C-tile layout == gate layout, no shuffles)
#pragma unroll
        for (int mt = 0; mt < 2; ++mt) {
#pragma unroll
            for (int v = 0; v < 8; ++v) {
                const int m = mt * 16 + v + hi8;
                const unsigned xoff =
                    ((unsigned)m * TLEN + (unsigned)tt) * N3U + jb + ln;
                const float xz = bf2f(xp[xoff]);
                const float xr = bf2f(xp[xoff + UNITS]);
                const float xh = bf2f(xp[xoff + 2 * UNITS]);
                const float z  = fast_sigmoid(xz + C[mt][0][v] + bz);
                const float r  = fast_sigmoid(xr + C[mt][1][v] + br);
                const float cd = fast_tanh(xh + r * (C[mt][2][v] + bh));
                const float hn = z * hprev[mt][v] + (1.f - z) * cd;
                hprev[mt][v] = hn;
                hbf[nxt][m * LDH + jb + ln] = f2bf(hn);
                const unsigned ooff =
                    ((unsigned)m * TLEN + (unsigned)tt) * (2 * UNITS) +
                    dir * UNITS + jb + ln;
                if (out_f32) ((float*)outp)[ooff] = hn;
                else         ((unsigned short*)outp)[ooff] = f2bf(hn);
            }
        }

        // prefetch next timestep's xp into cache (global_prefetch_b8)
        if (t + 1 < TLEN) {
            const int tn2 = dir ? (TLEN - 2 - t) : (t + 1);
            __builtin_prefetch(
                &xp[((unsigned)(ln + hi8) * TLEN + (unsigned)tn2) * N3U + jb],
                0, 0);
        }
        __syncthreads();
    }

    // final states
#pragma unroll
    for (int mt = 0; mt < 2; ++mt)
#pragma unroll
        for (int v = 0; v < 8; ++v) {
            const int m = mt * 16 + v + hi8;
            hfd[m * UNITS + jb + ln] = hprev[mt][v];
            if (out_states)
                out_states[m * (2 * UNITS) + dir * UNITS + jb + ln] =
                    hprev[mt][v];
        }
}

// ---------------------------------------------------------------------------
// Host orchestration
// ---------------------------------------------------------------------------
extern "C" void kernel_launch(void* const* d_in, const int* in_sizes, int n_in,
                              void* d_out, int out_size, void* d_ws,
                              size_t ws_size, hipStream_t stream) {
    (void)in_sizes; (void)n_in; (void)out_size; (void)ws_size;
    const float* x = (const float*)d_in[0];
    auto W = [&](int l, int d, int which) -> const float* {
        return (const float*)d_in[1 + l * 6 + d * 3 + which];
    };

    char* ws = (char*)d_ws;
    float* h_ping = (float*)(ws + 0);                       //  64 KB
    float* h_pong = (float*)(ws + 65536);                   //  64 KB
    unsigned short* wkt_fw = (unsigned short*)(ws + 131072);          // 768 KB
    unsigned short* wkt_bw = (unsigned short*)(ws + 131072 + 786432); // 768 KB
    unsigned short* wrt_fw = (unsigned short*)(ws + 1703936);         // 384 KB
    unsigned short* wrt_bw = (unsigned short*)(ws + 2097152);         // 384 KB
    unsigned short* xp_fw  = (unsigned short*)(ws + 2490368);         //  96 MB
    unsigned short* xp_bw  = xp_fw + (size_t)MROWS * N3U;             //  96 MB
    unsigned short* io0    = (unsigned short*)(ws + 203816960);       //  64 MB
    unsigned short* io1    = io0 + (size_t)MROWS * 2 * UNITS;         //  64 MB
    float* outp = (float*)d_out;

    const int nproj = MROWS * N3U;            // 50331648
    const int ZB = 256;

    // h0 for layer 0 is zeros
    zero_f32_kernel<<<(2 * BATCH * UNITS + ZB - 1) / ZB, ZB, 0, stream>>>(
        h_ping, 2 * BATCH * UNITS);

    // ---------------- layer 0 ----------------
    transpose_bf16_kernel<<<(N3U * UNITS + ZB - 1) / ZB, ZB, 0, stream>>>(
        W(0, 0, 1), wrt_fw, UNITS);
    transpose_bf16_kernel<<<(N3U * UNITS + ZB - 1) / ZB, ZB, 0, stream>>>(
        W(0, 1, 1), wrt_bw, UNITS);
    proj0_kernel<<<(nproj + ZB - 1) / ZB, ZB, 0, stream>>>(
        x, W(0, 0, 0), W(0, 0, 2), W(0, 1, 0), W(0, 1, 2), xp_fw, xp_bw);
    gru_scan_kernel<<<2, 512, 0, stream>>>(
        xp_fw, xp_bw, wrt_fw, wrt_bw, W(0, 0, 2) + N3U, W(0, 1, 2) + N3U,
        h_ping, h_pong, (void*)io0, 0, nullptr);

    // ---------------- layer 1 ----------------
    transpose_bf16_kernel<<<(N3U * 512 + ZB - 1) / ZB, ZB, 0, stream>>>(
        W(1, 0, 0), wkt_fw, 512);
    transpose_bf16_kernel<<<(N3U * 512 + ZB - 1) / ZB, ZB, 0, stream>>>(
        W(1, 1, 0), wkt_bw, 512);
    transpose_bf16_kernel<<<(N3U * UNITS + ZB - 1) / ZB, ZB, 0, stream>>>(
        W(1, 0, 1), wrt_fw, UNITS);
    transpose_bf16_kernel<<<(N3U * UNITS + ZB - 1) / ZB, ZB, 0, stream>>>(
        W(1, 1, 1), wrt_bw, UNITS);
    proj_gemm_kernel<<<6144, 256, 0, stream>>>(io0, wkt_fw, W(1, 0, 2), xp_fw);
    proj_gemm_kernel<<<6144, 256, 0, stream>>>(io0, wkt_bw, W(1, 1, 2), xp_bw);
    gru_scan_kernel<<<2, 512, 0, stream>>>(
        xp_fw, xp_bw, wrt_fw, wrt_bw, W(1, 0, 2) + N3U, W(1, 1, 2) + N3U,
        h_pong, h_ping, (void*)io1, 0, nullptr);

    // ---------------- layer 2 ----------------
    transpose_bf16_kernel<<<(N3U * 512 + ZB - 1) / ZB, ZB, 0, stream>>>(
        W(2, 0, 0), wkt_fw, 512);
    transpose_bf16_kernel<<<(N3U * 512 + ZB - 1) / ZB, ZB, 0, stream>>>(
        W(2, 1, 0), wkt_bw, 512);
    transpose_bf16_kernel<<<(N3U * UNITS + ZB - 1) / ZB, ZB, 0, stream>>>(
        W(2, 0, 1), wrt_fw, UNITS);
    transpose_bf16_kernel<<<(N3U * UNITS + ZB - 1) / ZB, ZB, 0, stream>>>(
        W(2, 1, 1), wrt_bw, UNITS);
    proj_gemm_kernel<<<6144, 256, 0, stream>>>(io1, wkt_fw, W(2, 0, 2), xp_fw);
    proj_gemm_kernel<<<6144, 256, 0, stream>>>(io1, wkt_bw, W(2, 1, 2), xp_bw);
    gru_scan_kernel<<<2, 512, 0, stream>>>(
        xp_fw, xp_bw, wrt_fw, wrt_bw, W(2, 0, 2) + N3U, W(2, 1, 2) + N3U,
        h_ping, h_pong, (void*)outp, 1, outp + OUT_MAIN);
}
